// RobustDecoder_17068200035049
// MI455X (gfx1250) — compile-verified, matrix operand order checked by reading, not run on hardware
//
#include <hip/hip_runtime.h>
#include <hip/hip_bf16.h>

// ---------------------------------------------------------------------------
// Model dims (fixed by the reference)
// ---------------------------------------------------------------------------
#define BATCH    8
#define D_MODEL  256
#define D_INNER  512
#define D_STATE  16
#define D_CONV   4
#define DT_RANK  16
#define N_LAYERS 4
#define N_DEPTHS 4
#define N_EMB    8192
#define SEQ_L    1024      // 32*32 tokens after the conv stem

#define USE_ASYNC_COPY 1   // gfx1250 global_load_async_to_lds_b128 path

typedef __attribute__((ext_vector_type(16))) _Float16 v16h;
typedef __attribute__((ext_vector_type(8)))  float    v8f;

__device__ __forceinline__ float silu_f(float v) {
    return v * (1.0f / (1.0f + __expf(-v)));
}

#define EP_NONE     0
#define EP_SOFTPLUS 1

// ---------------------------------------------------------------------------
// Raw-run loader: per ISA 16-bit A/B 16x32 layout one lane owns two
// contiguous 8-float runs: K = kb + half*8 + [0..7] and kb+16+half*8 + [0..7].
// Each run = 2x float4 (global_load_b128). K is compile time: for K=16 the
// second run is statically zero (dt_proj); otherwise unguarded.
// ---------------------------------------------------------------------------
template<int K>
__device__ __forceinline__ void load_runs(const float* __restrict__ row,
                                          int kb, int half,
                                          float4& lo0, float4& lo1,
                                          float4& hi0, float4& hi1)
{
    const float4* p1 = reinterpret_cast<const float4*>(row + kb + half * 8);
    lo0 = p1[0];
    lo1 = p1[1];
    if ((K & 31) == 0 || kb + 32 <= K) {
        const float4* p2 =
            reinterpret_cast<const float4*>(row + kb + 16 + half * 8);
        hi0 = p2[0];
        hi1 = p2[1];
    } else {
        hi0 = make_float4(0.f, 0.f, 0.f, 0.f);
        hi1 = make_float4(0.f, 0.f, 0.f, 0.f);
    }
}

__device__ __forceinline__ v16h cvt_frag(float4 a, float4 b, float4 c, float4 d)
{
    v16h v;
    v[0]  = (_Float16)a.x; v[1]  = (_Float16)a.y;
    v[2]  = (_Float16)a.z; v[3]  = (_Float16)a.w;
    v[4]  = (_Float16)b.x; v[5]  = (_Float16)b.y;
    v[6]  = (_Float16)b.z; v[7]  = (_Float16)b.w;
    v[8]  = (_Float16)c.x; v[9]  = (_Float16)c.y;
    v[10] = (_Float16)c.z; v[11] = (_Float16)c.w;
    v[12] = (_Float16)d.x; v[13] = (_Float16)d.y;
    v[14] = (_Float16)d.z; v[15] = (_Float16)d.w;
    return v;
}

// ---------------------------------------------------------------------------
// WMMA GEMM:  C[M,N] = A[M,K(lda)] * W[N,K]^T (+bias) (+epi) (+residual)
// One wave computes a 64x16 strip (4 M-tiles sharing one B fragment).
// All 20 b128 loads of a K-step are issued before any conversion so the
// wave pays a single s_wait_loadcnt per iteration, then runs 4 independent
// back-to-back v_wmma. Requires M%64==0, N%16==0 (true at every call site).
// ---------------------------------------------------------------------------
template<int K>
__global__ void __launch_bounds__(128, 1)
gemm_wmma_f16(const float* __restrict__ A, int lda,
              const float* __restrict__ W,          // N x K
              const float* __restrict__ bias,       // N or null
              const float* __restrict__ residual,   // M x N or null
              float* __restrict__ C,
              int M, int N, int epilogue)
{
    const int lane   = threadIdx.x & 31;
    const int wave   = threadIdx.x >> 5;
    const int tilesN = N >> 4;
    const int strips = (M >> 6) * tilesN;
    const int strip  = blockIdx.x * (blockDim.x >> 5) + wave;  // wave-uniform
    if (strip >= strips) return;                               // whole-wave exit
    const int sm = strip / tilesN;
    const int tn = strip - sm * tilesN;

    const int r    = lane & 15;
    const int half = lane >> 4;

    const float* Ar0  = A + (size_t)(sm * 64 + r) * lda;
    const float* Ar1  = Ar0 + (size_t)16 * lda;
    const float* Ar2  = Ar0 + (size_t)32 * lda;
    const float* Ar3  = Ar0 + (size_t)48 * lda;
    const float* Wrow = W + (size_t)(tn * 16 + r) * K;

    v8f c0 = {}, c1 = {}, c2 = {}, c3 = {};
    for (int kb = 0; kb < K; kb += 32) {
        // --- issue all loads first (independent destinations) ---
        float4 wb0, wb1, wb2, wb3;
        float4 a00, a01, a02, a03;
        float4 a10, a11, a12, a13;
        float4 a20, a21, a22, a23;
        float4 a30, a31, a32, a33;
        load_runs<K>(Wrow, kb, half, wb0, wb1, wb2, wb3);
        load_runs<K>(Ar0,  kb, half, a00, a01, a02, a03);
        load_runs<K>(Ar1,  kb, half, a10, a11, a12, a13);
        load_runs<K>(Ar2,  kb, half, a20, a21, a22, a23);
        load_runs<K>(Ar3,  kb, half, a30, a31, a32, a33);
        // --- convert ---
        const v16h fb = cvt_frag(wb0, wb1, wb2, wb3);
        const v16h f0 = cvt_frag(a00, a01, a02, a03);
        const v16h f1 = cvt_frag(a10, a11, a12, a13);
        const v16h f2 = cvt_frag(a20, a21, a22, a23);
        const v16h f3 = cvt_frag(a30, a31, a32, a33);
        // --- 4 independent WMMAs ---
        c0 = __builtin_amdgcn_wmma_f32_16x16x32_f16(false, f0, false, fb,
                                                    (short)0, c0, false, false);
        c1 = __builtin_amdgcn_wmma_f32_16x16x32_f16(false, f1, false, fb,
                                                    (short)0, c1, false, false);
        c2 = __builtin_amdgcn_wmma_f32_16x16x32_f16(false, f2, false, fb,
                                                    (short)0, c2, false, false);
        c3 = __builtin_amdgcn_wmma_f32_16x16x32_f16(false, f3, false, fb,
                                                    (short)0, c3, false, false);
    }

    // C layout: lanes 0-15 -> M = base + vgpr, lanes 16-31 -> M = base+8+vgpr
    const int   n  = tn * 16 + r;
    const float bv = bias ? bias[n] : 0.0f;
    v8f acc[4] = {c0, c1, c2, c3};
#pragma unroll
    for (int t = 0; t < 4; ++t) {
        const int m0 = sm * 64 + t * 16;
#pragma unroll
        for (int rr = 0; rr < 8; ++rr) {
            const int m = m0 + rr + half * 8;
            float v = acc[t][rr] + bv;
            if (epilogue == EP_SOFTPLUS)
                v = (v > 20.0f) ? v : log1pf(__expf(v));
            if (residual) v += residual[(size_t)m * N + n];
            C[(size_t)m * N + n] = v;
        }
    }
}

// ---------------------------------------------------------------------------
// Conv stem
// ---------------------------------------------------------------------------
__global__ void conv2d_s2_kernel(const float* __restrict__ in,
                                 const float* __restrict__ w,
                                 const float* __restrict__ bias,
                                 float* __restrict__ out,
                                 int B, int Cin, int Hin, int Win,
                                 int Cout, int Hout, int Wout)
{
    const int total = B * Cout * Hout * Wout;
    int idx = blockIdx.x * blockDim.x + threadIdx.x;
    if (idx >= total) return;
    int ox = idx % Wout; int t = idx / Wout;
    int oy = t % Hout;   t /= Hout;
    int oc = t % Cout;   int b = t / Cout;
    float acc = bias[oc];
    for (int ic = 0; ic < Cin; ++ic) {
        const float* ip = in + ((size_t)b * Cin + ic) * Hin * Win;
        const float* wp = w + ((size_t)oc * Cin + ic) * 9;
#pragma unroll
        for (int ky = 0; ky < 3; ++ky) {
            const int iy = oy * 2 + ky - 1;
            if (iy < 0 || iy >= Hin) continue;
#pragma unroll
            for (int kx = 0; kx < 3; ++kx) {
                const int ix = ox * 2 + kx - 1;
                if (ix < 0 || ix >= Win) continue;
                acc += ip[(size_t)iy * Win + ix] * wp[ky * 3 + kx];
            }
        }
    }
    out[idx] = acc;
}

// GroupNorm + SiLU, in place. One block per (batch, group).
__global__ void groupnorm_silu_kernel(float* __restrict__ x,
                                      const float* __restrict__ g,
                                      const float* __restrict__ bta,
                                      int C, int HW, int Cg)
{
    const int groups = C / Cg;
    const int b = blockIdx.x / groups;
    const int gr = blockIdx.x % groups;
    const size_t base = ((size_t)b * C + (size_t)gr * Cg) * HW;
    const int n = Cg * HW;
    __shared__ float s0[256], s1[256];
    float sum = 0.0f, sq = 0.0f;
    for (int i = threadIdx.x; i < n; i += blockDim.x) {
        const float v = x[base + i];
        sum += v; sq += v * v;
    }
    s0[threadIdx.x] = sum; s1[threadIdx.x] = sq;
    __syncthreads();
    for (int s = 128; s > 0; s >>= 1) {
        if ((int)threadIdx.x < s) {
            s0[threadIdx.x] += s0[threadIdx.x + s];
            s1[threadIdx.x] += s1[threadIdx.x + s];
        }
        __syncthreads();
    }
    const float m  = s0[0] / (float)n;
    const float rs = rsqrtf(s1[0] / (float)n - m * m + 1e-5f);
    for (int i = threadIdx.x; i < n; i += blockDim.x) {
        const int ch = gr * Cg + i / HW;
        const float v = (x[base + i] - m) * rs * g[ch] + bta[ch];
        x[base + i] = silu_f(v);
    }
}

// (B,C,L) -> (B,L,C)
__global__ void chw_to_lc_kernel(const float* __restrict__ h,
                                 float* __restrict__ f)
{
    int idx = blockIdx.x * blockDim.x + threadIdx.x;
    if (idx >= BATCH * SEQ_L * D_MODEL) return;
    const int c = idx & (D_MODEL - 1);
    const int l = (idx >> 8) & (SEQ_L - 1);
    const int b = idx >> 18;
    f[idx] = h[((size_t)b * D_MODEL + c) * SEQ_L + l];
}

// LayerNorm over last dim (256); one block per token.
__global__ void layernorm_kernel(const float* __restrict__ in,
                                 const float* __restrict__ g,
                                 const float* __restrict__ b,
                                 float* __restrict__ out)
{
    const int token = blockIdx.x;
    const int tid = threadIdx.x;
    const float v = in[(size_t)token * D_MODEL + tid];
    __shared__ float s0[256], s1[256];
    s0[tid] = v; s1[tid] = v * v;
    __syncthreads();
    for (int s = 128; s > 0; s >>= 1) {
        if (tid < s) { s0[tid] += s0[tid + s]; s1[tid] += s1[tid + s]; }
        __syncthreads();
    }
    const float m  = s0[0] * (1.0f / D_MODEL);
    const float rs = rsqrtf(s1[0] * (1.0f / D_MODEL) - m * m + 1e-5f);
    out[(size_t)token * D_MODEL + tid] = (v - m) * rs * g[tid] + b[tid];
}

// Causal depthwise conv1d over L (+bias +SiLU). Reads xc half of xz.
__global__ void dwconv_silu_kernel(const float* __restrict__ xz, // (B,L,2*Di)
                                   const float* __restrict__ w,  // (Di,4)
                                   const float* __restrict__ cb, // (Di)
                                   float* __restrict__ out)      // (B,L,Di)
{
    int idx = blockIdx.x * blockDim.x + threadIdx.x;
    if (idx >= BATCH * SEQ_L * D_INNER) return;
    const int d = idx & (D_INNER - 1);
    const int t = (idx >> 9) & (SEQ_L - 1);
    const int b = idx >> 19;
    float acc = cb[d];
#pragma unroll
    for (int j = 0; j < D_CONV; ++j) {
        const int tt = t + j - (D_CONV - 1);
        if (tt >= 0)
            acc += w[d * D_CONV + j] *
                   xz[((size_t)b * SEQ_L + tt) * (2 * D_INNER) + d];
    }
    out[idx] = silu_f(acc);
}

// ---------------------------------------------------------------------------
// Selective scan. One block per batch (512 threads = all d_inner channels);
// B_t/C_t (32 floats per step, shared by all channels) are staged into LDS
// in 64-step chunks with gfx1250 async global->LDS copies (ASYNCcnt), then
// broadcast-read from LDS. 16-state recurrence lives in registers.
// ---------------------------------------------------------------------------
#define SCAN_CHUNK 64

__global__ void __launch_bounds__(D_INNER)
selective_scan_kernel(const float* __restrict__ dtb,   // (B,L,Di)
                      const float* __restrict__ xdbl,  // (B,L,48)
                      const float* __restrict__ xcc,   // (B,L,Di)
                      const float* __restrict__ A_log, // (Di,16)
                      float* __restrict__ y)           // (B,L,Di)
{
    const int b = blockIdx.x;          // BATCH blocks
    const int d = threadIdx.x;         // D_INNER threads

    __shared__ float sBC[SCAN_CHUNK * 2 * D_STATE];   // 8 KB: [step][B(16)|C(16)]

    float Aneg[D_STATE], h[D_STATE];
#pragma unroll
    for (int s = 0; s < D_STATE; ++s) {
        Aneg[s] = -__expf(A_log[d * D_STATE + s]);
        h[s] = 0.0f;
    }

    const size_t rowB = (size_t)b * SEQ_L;
    for (int t0 = 0; t0 < SEQ_L; t0 += SCAN_CHUNK) {
        __syncthreads();   // previous chunk fully consumed before overwrite
        {
            // thread i copies 16 bytes: row rr = i/8, quarter p = i%8 of the
            // 32-float (B|C) slice at xdbl[row, 16:48]; 16B aligned both sides.
            const int rr = threadIdx.x >> 3;
            const int p  = threadIdx.x & 7;
            const float* src =
                xdbl + (rowB + t0 + rr) * (DT_RANK + 2 * D_STATE) + DT_RANK + p * 4;
            float* dst = &sBC[rr * 2 * D_STATE + p * 4];
#if USE_ASYNC_COPY
            unsigned lds_off = (unsigned)(size_t)dst;   // LDS byte address
            asm volatile("global_load_async_to_lds_b128 %0, %1, off"
                         :: "v"(lds_off), "v"(src) : "memory");
#else
            *reinterpret_cast<float4*>(dst) =
                *reinterpret_cast<const float4*>(src);
#endif
        }
#if USE_ASYNC_COPY
        asm volatile("s_wait_asynccnt 0x0" ::: "memory");
#endif
        __syncthreads();

        for (int tt = 0; tt < SCAN_CHUNK; ++tt) {
            const size_t row = rowB + t0 + tt;
            const float dt = dtb[row * D_INNER + d];
            const float xv = xcc[row * D_INNER + d];
            const float dx = dt * xv;
            const float* Bt = &sBC[tt * 2 * D_STATE];
            const float* Ct = Bt + D_STATE;
            float acc = 0.0f;
#pragma unroll
            for (int s = 0; s < D_STATE; ++s) {
                const float dA = __expf(dt * Aneg[s]);
                h[s] = dA * h[s] + dx * Bt[s];
                acc += h[s] * Ct[s];
            }
            y[row * D_INNER + d] = acc;
        }
    }
}

// y = (y + Dp[d]*xc) * silu(z), in place on y.
__global__ void gate_kernel(float* __restrict__ y,
                            const float* __restrict__ xcc,
                            const float* __restrict__ xz,
                            const float* __restrict__ Dp)
{
    int idx = blockIdx.x * blockDim.x + threadIdx.x;
    if (idx >= BATCH * SEQ_L * D_INNER) return;
    const int d = idx & (D_INNER - 1);
    const size_t bt = (size_t)(idx >> 9);
    const float z = xz[bt * (2 * D_INNER) + D_INNER + d];
    y[idx] = (y[idx] + Dp[d] * xcc[idx]) * silu_f(z);
}

// Adaptive avg pool 32x32 -> 8x8 (4x4 windows); f (B,L,C) -> feat (B,64,C)
__global__ void pool_kernel(const float* __restrict__ f,
                            float* __restrict__ feat)
{
    int idx = blockIdx.x * blockDim.x + threadIdx.x;
    if (idx >= BATCH * 64 * D_MODEL) return;
    const int c = idx & (D_MODEL - 1);
    const int p = (idx >> 8) & 63;
    const int b = idx >> 14;
    const int ph = p >> 3, pw = p & 7;
    float acc = 0.0f;
#pragma unroll
    for (int dy = 0; dy < 4; ++dy)
#pragma unroll
        for (int dx = 0; dx < 4; ++dx) {
            const int l = (ph * 4 + dy) * 32 + (pw * 4 + dx);
            acc += f[((size_t)b * SEQ_L + l) * D_MODEL + c];
        }
    feat[idx] = acc * (1.0f / 16.0f);
}

// ---------------------------------------------------------------------------
// Launcher
// ---------------------------------------------------------------------------
static inline int cdiv(int a, int b) { return (a + b - 1) / b; }

template<int K>
static void launch_gemm(const float* A, int lda, const float* W,
                        const float* bias, const float* residual, float* C,
                        int M, int N, int epi, hipStream_t stream)
{
    const int strips = (M >> 6) * (N >> 4);     // 64x16 strip per wave
    const int wavesPerBlock = 4;                // 128 threads
    const int blocks = cdiv(strips, wavesPerBlock);
    gemm_wmma_f16<K><<<blocks, wavesPerBlock * 32, 0, stream>>>(
        A, lda, W, bias, residual, C, M, N, epi);
}

extern "C" void kernel_launch(void* const* d_in, const int* in_sizes, int n_in,
                              void* d_out, int out_size, void* d_ws, size_t ws_size,
                              hipStream_t stream)
{
    (void)in_sizes; (void)n_in; (void)out_size; (void)ws_size;
    const float* x       = (const float*)d_in[0];
    const float* conv1_w = (const float*)d_in[1];
    const float* conv1_b = (const float*)d_in[2];
    const float* gn1_g   = (const float*)d_in[3];
    const float* gn1_b   = (const float*)d_in[4];
    const float* conv2_w = (const float*)d_in[5];
    const float* conv2_b = (const float*)d_in[6];
    const float* gn2_g   = (const float*)d_in[7];
    const float* gn2_b   = (const float*)d_in[8];
    const float* ln_g    = (const float*)d_in[9];
    const float* ln_b    = (const float*)d_in[10];
    const float* in_w    = (const float*)d_in[11];
    const float* cw      = (const float*)d_in[12];
    const float* cb      = (const float*)d_in[13];
    const float* xp_w    = (const float*)d_in[14];
    const float* dtp_w   = (const float*)d_in[15];
    const float* dtp_b   = (const float*)d_in[16];
    const float* A_log   = (const float*)d_in[17];
    const float* Dp      = (const float*)d_in[18];
    const float* out_w   = (const float*)d_in[19];
    const float* head_w  = (const float*)d_in[20];
    const float* head_b  = (const float*)d_in[21];
    float* out = (float*)d_out;

    // Workspace layout (bytes), with region reuse. Everything stays L2-warm.
    char* ws = (char*)d_ws;
    float* h1   = (float*)(ws);                           // 16 MB (conv1 out)
    float* h2   = (float*)(ws + (size_t)16 * (1 << 20));  //  8 MB (conv2 out)
    float* f    = (float*)(ws + (size_t)24 * (1 << 20));  //  8 MB (residual stream)
    float* lnb  = (float*)(ws + (size_t)32 * (1 << 20));  //  8 MB (layernorm out)
    float* xz   = (float*)(ws + (size_t)40 * (1 << 20));  // 32 MB (in_proj out)
    float* xcc  = (float*)(ws + (size_t)72 * (1 << 20));  // 16 MB (dw-conv out)
    float* ybuf = (float*)(ws + (size_t)88 * (1 << 20));  // 16 MB (scan/gated out)
    float* dtb  = h1;    // reuse: h1 dead after conv2
    float* xdbl = h2;    // reuse: h2 dead after transpose
    float* feat = lnb;   // reuse: layernorm buffer dead after layers

    const int BL = BATCH * SEQ_L;   // 8192 tokens

    // ---- conv stem ----
    {
        const int total1 = BATCH * 128 * 64 * 64;
        conv2d_s2_kernel<<<cdiv(total1, 256), 256, 0, stream>>>(
            x, conv1_w, conv1_b, h1, BATCH, 3, 128, 128, 128, 64, 64);
        groupnorm_silu_kernel<<<BATCH * 8, 256, 0, stream>>>(
            h1, gn1_g, gn1_b, 128, 64 * 64, 16);

        const int total2 = BATCH * 256 * 32 * 32;
        conv2d_s2_kernel<<<cdiv(total2, 256), 256, 0, stream>>>(
            h1, conv2_w, conv2_b, h2, BATCH, 128, 64, 64, 256, 32, 32);
        groupnorm_silu_kernel<<<BATCH * 8, 256, 0, stream>>>(
            h2, gn2_g, gn2_b, 256, 32 * 32, 32);

        chw_to_lc_kernel<<<cdiv(BL * D_MODEL, 256), 256, 0, stream>>>(h2, f);
    }

    // ---- Mamba layers ----
    for (int i = 0; i < N_LAYERS; ++i) {
        const float* in_w_i  = in_w  + (size_t)i * 2 * D_INNER * D_MODEL;
        const float* cw_i    = cw    + (size_t)i * D_INNER * D_CONV;
        const float* cb_i    = cb    + (size_t)i * D_INNER;
        const float* xp_w_i  = xp_w  + (size_t)i * (DT_RANK + 2 * D_STATE) * D_INNER;
        const float* dtp_w_i = dtp_w + (size_t)i * D_INNER * DT_RANK;
        const float* dtp_b_i = dtp_b + (size_t)i * D_INNER;
        const float* A_log_i = A_log + (size_t)i * D_INNER * D_STATE;
        const float* Dp_i    = Dp    + (size_t)i * D_INNER;
        const float* out_w_i = out_w + (size_t)i * D_MODEL * D_INNER;

        layernorm_kernel<<<BL, D_MODEL, 0, stream>>>(
            f, ln_g + (size_t)i * D_MODEL, ln_b + (size_t)i * D_MODEL, lnb);

        // in_proj: (8192 x 256) x (1024 x 256)^T -> xz
        launch_gemm<D_MODEL>(lnb, D_MODEL, in_w_i, nullptr, nullptr, xz,
                             BL, 2 * D_INNER, EP_NONE, stream);

        dwconv_silu_kernel<<<cdiv(BL * D_INNER, 256), 256, 0, stream>>>(
            xz, cw_i, cb_i, xcc);

        // x_proj: (8192 x 512) x (48 x 512)^T -> xdbl
        launch_gemm<D_INNER>(xcc, D_INNER, xp_w_i, nullptr, nullptr, xdbl,
                             BL, DT_RANK + 2 * D_STATE, EP_NONE, stream);

        // dt_proj + softplus: A = xdbl[:, :16] (lda=48), K=16
        launch_gemm<DT_RANK>(xdbl, DT_RANK + 2 * D_STATE, dtp_w_i, dtp_b_i,
                             nullptr, dtb, BL, D_INNER, EP_SOFTPLUS, stream);

        selective_scan_kernel<<<BATCH, D_INNER, 0, stream>>>(
            dtb, xdbl, xcc, A_log_i, ybuf);

        gate_kernel<<<cdiv(BL * D_INNER, 256), 256, 0, stream>>>(
            ybuf, xcc, xz, Dp_i);

        // out_proj + residual: f += gated @ out_w^T
        launch_gemm<D_INNER>(ybuf, D_INNER, out_w_i, nullptr, f, f,
                             BL, D_MODEL, EP_NONE, stream);
    }

    // ---- pool + heads ----
    pool_kernel<<<cdiv(BATCH * 64 * D_MODEL, 256), 256, 0, stream>>>(f, feat);

    for (int d = 0; d < N_DEPTHS; ++d) {
        launch_gemm<D_MODEL>(feat, D_MODEL,
                             head_w + (size_t)d * N_EMB * D_MODEL,
                             head_b + (size_t)d * N_EMB,
                             nullptr,
                             out + (size_t)d * BATCH * 64 * N_EMB,
                             BATCH * 64, N_EMB, EP_NONE, stream);
    }
}
